// TransformerEncoderBlock_42150809043174
// MI455X (gfx1250) — matrix operand order repaired and checked by reading, not compiled, not run on hardware
//
#include <hip/hip_runtime.h>
#include <hip/hip_bf16.h>

// ---------------------------------------------------------------------------
// Transformer encoder block (pre-LN causal MHA + FFN) for gfx1250 (MI455X).
// T=2048, D=1024, H=16, HS=64, DFF=4096.
// All matmuls via v_wmma_f32_16x16x32_bf16 (bf16 operands, f32 accumulate).
// Weights pre-converted+transposed to bf16 [N x K] -> every WMMA operand is a
// pair of contiguous b128 loads.  GEMM inner loops are software-pipelined with
// explicit ping-pong register buffers (unroll-by-2 in source; compiler
// unrolling disabled so the back-edge needs no register copies).
// Attention is streaming (flash) softmax.
// ---------------------------------------------------------------------------

typedef __attribute__((ext_vector_type(16))) __bf16 v16bf;
typedef __attribute__((ext_vector_type(8)))  float  v8f;

constexpr int T   = 2048;
constexpr int D   = 1024;
constexpr int H   = 16;
constexpr int HS  = 64;
constexpr int DFF = 4096;

struct bf16x32 { uint4 lo, hi; };   // 32 bytes == v16bf

__device__ __forceinline__ __bf16 f2bf(float f) {
  unsigned u = __builtin_bit_cast(unsigned, f);
  u += 0x7FFFu + ((u >> 16) & 1u);               // round-to-nearest-even
  unsigned short h = (unsigned short)(u >> 16);
  return __builtin_bit_cast(__bf16, h);
}

__device__ __forceinline__ v8f wmma_bf16(v16bf a, v16bf b, v8f c) {
  return __builtin_amdgcn_wmma_f32_16x16x32_bf16(false, a, false, b, (short)0, c,
                                                 false, false);
}

// A operand: row-major bf16 [M x K]; 16x32 tile at (rowBase, k0).
// Lane L<16: row L, K = k0+0..7 | k0+16..23 ; lane L>=16: row L-16, +8.. / +24..
__device__ __forceinline__ v16bf load_a_bf16(const __bf16* A, int lda, int rowBase,
                                             int k0, int lane) {
  const __bf16* p = A + (size_t)(rowBase + (lane & 15)) * lda +
                    (k0 + ((lane >> 4) & 1) * 8);
  bf16x32 r{*(const uint4*)p, *(const uint4*)(p + 16)};
  return __builtin_bit_cast(v16bf, r);
}

// B operand from TRANSPOSED bf16 [N x K] (contiguous along K).
// Lane L<16: col L, K = k0+0..7 | k0+16..23 ; lane L>=16: col L-16, +8.. / +24..
__device__ __forceinline__ v16bf load_bT_bf16(const __bf16* Bt, int ldb, int k0,
                                              int colBase, int lane) {
  const __bf16* p = Bt + (size_t)(colBase + (lane & 15)) * ldb +
                    (k0 + ((lane >> 4) & 1) * 8);
  bf16x32 r{*(const uint4*)p, *(const uint4*)(p + 16)};
  return __builtin_bit_cast(v16bf, r);
}

// Software-pipelined 16(M) x 64(N) x KDIM GEMM core: c[4] += A[16xK] * Bt^T.
// Explicit ping-pong double buffer, unrolled by 2 in source.  Compiler loop
// unrolling is disabled: the back-edge maps set0->set0 / set1->set1, so no
// rotation copies (and no induced WMMA hazard NOPs) are needed.
template <int KDIM>
__device__ __forceinline__ void gemm16x64(const __bf16* __restrict__ A, int lda,
                                          int rowBase,
                                          const __bf16* __restrict__ Bt, int ldb,
                                          int colBase, int lane, v8f c[4]) {
  static_assert(KDIM % 64 == 0 && KDIM >= 128, "KDIM must be multiple of 64");
  const __bf16* pa = A + (size_t)(rowBase + (lane & 15)) * lda;
  v16bf a0 = load_a_bf16(A, lda, rowBase, 0, lane);
  v16bf b0[4];
#pragma unroll
  for (int j = 0; j < 4; ++j)
    b0[j] = load_bT_bf16(Bt, ldb, 0, colBase + j * 16, lane);
  v16bf a1;
  v16bf b1[4];
#pragma unroll 1
  for (int k0 = 0; k0 < KDIM - 64; k0 += 64) {
    a1 = load_a_bf16(A, lda, rowBase, k0 + 32, lane);
#pragma unroll
    for (int j = 0; j < 4; ++j)
      b1[j] = load_bT_bf16(Bt, ldb, k0 + 32, colBase + j * 16, lane);
    __builtin_prefetch(pa + k0 + 192, 0, 1);
#pragma unroll
    for (int j = 0; j < 4; ++j) c[j] = wmma_bf16(a0, b0[j], c[j]);
    a0 = load_a_bf16(A, lda, rowBase, k0 + 64, lane);
#pragma unroll
    for (int j = 0; j < 4; ++j)
      b0[j] = load_bT_bf16(Bt, ldb, k0 + 64, colBase + j * 16, lane);
#pragma unroll
    for (int j = 0; j < 4; ++j) c[j] = wmma_bf16(a1, b1[j], c[j]);
  }
  // tail: set0 holds K = KDIM-64; load and apply KDIM-32.
  a1 = load_a_bf16(A, lda, rowBase, KDIM - 32, lane);
#pragma unroll
  for (int j = 0; j < 4; ++j)
    b1[j] = load_bT_bf16(Bt, ldb, KDIM - 32, colBase + j * 16, lane);
#pragma unroll
  for (int j = 0; j < 4; ++j) c[j] = wmma_bf16(a0, b0[j], c[j]);
#pragma unroll
  for (int j = 0; j < 4; ++j) c[j] = wmma_bf16(a1, b1[j], c[j]);
}

// ----------------- weight convert + transpose prep kernels -----------------
// QKV weights [mat][H][D][HS] fp32 -> [mat][H][HS][D] bf16 (one flat buffer)
__global__ __launch_bounds__(256) void wcvt_qkv_kernel(const float* __restrict__ Wq,
    const float* __restrict__ Wk, const float* __restrict__ Wv,
    __bf16* __restrict__ out) {
  size_t i = (size_t)blockIdx.x * 256 + threadIdx.x;   // 3*H*HS*D elements
  const int d = (int)(i & (D - 1));
  size_t r = i >> 10;
  const int hs   = (int)(r & (HS - 1)); r >>= 6;
  const int head = (int)(r & (H - 1));
  const int mat  = (int)(r >> 4);
  const float* W = mat == 0 ? Wq : mat == 1 ? Wk : Wv;
  out[i] = f2bf(W[((size_t)head * D + d) * HS + hs]);
}

// Generic: W row-major [K x N] fp32 -> out [N x K] bf16, K = 1<<kbits
__global__ __launch_bounds__(256) void wcvt_t_kernel(const float* __restrict__ W,
    __bf16* __restrict__ out, int N, int kbits) {
  size_t i = (size_t)blockIdx.x * 256 + threadIdx.x;   // N*K elements
  const int K = 1 << kbits;
  const int k = (int)(i & (K - 1));
  const int n = (int)(i >> kbits);
  out[i] = f2bf(W[(size_t)k * N + n]);
}

// --------------------------- LayerNorm -> bf16 -----------------------------
__global__ __launch_bounds__(256) void ln_kernel(const float* __restrict__ x,
                                                 const float* __restrict__ sc,
                                                 const float* __restrict__ bi,
                                                 __bf16* __restrict__ out) {
  __shared__ float red[256];
  const int row = blockIdx.x, tid = threadIdx.x;
  const float* xr = x + (size_t)row * D;
  float v[4];
  float s = 0.f;
#pragma unroll
  for (int j = 0; j < 4; ++j) { v[j] = xr[tid + j * 256]; s += v[j]; }
  red[tid] = s; __syncthreads();
  for (int o = 128; o > 0; o >>= 1) { if (tid < o) red[tid] += red[tid + o]; __syncthreads(); }
  const float mu = red[0] * (1.f / D);
  __syncthreads();
  float vs = 0.f;
#pragma unroll
  for (int j = 0; j < 4; ++j) { float d = v[j] - mu; vs += d * d; }
  red[tid] = vs; __syncthreads();
  for (int o = 128; o > 0; o >>= 1) { if (tid < o) red[tid] += red[tid + o]; __syncthreads(); }
  const float rstd = rsqrtf(red[0] * (1.f / D) + 1e-6f);
  __bf16* orow = out + (size_t)row * D;
#pragma unroll
  for (int j = 0; j < 4; ++j) {
    const int c = tid + j * 256;
    orow[c] = f2bf((v[j] - mu) * rstd * sc[c] + bi[c]);
  }
}

// ------------------------------- QKV GEMMs ---------------------------------
// One wave computes a 16(M) x 64(N=HS) tile for one (mat, head, tTile).
// wave id = (mat*H + head)*(T/16) + tTile   -> 3*16*128 = 6144 waves
// V is stored TRANSPOSED [H][HS][T] for the attention P*V operand.
__global__ __launch_bounds__(256) void qkv_kernel(const __bf16* __restrict__ hA,
    const __bf16* __restrict__ WT,   // [mat][head][HS][D] bf16
    __bf16* __restrict__ q, __bf16* __restrict__ k, __bf16* __restrict__ vt) {
  const int lane = threadIdx.x & 31;
  int gw = (blockIdx.x * blockDim.x + threadIdx.x) >> 5;
  const int tTile = gw & 127;  gw >>= 7;
  const int head  = gw & 15;
  const int mat   = gw >> 4;
  const __bf16* W = WT + ((size_t)(mat * H + head)) * HS * D;
  v8f c[4];
#pragma unroll
  for (int j = 0; j < 4; ++j)
#pragma unroll
    for (int i = 0; i < 8; ++i) c[j][i] = 0.f;

  gemm16x64<D>(hA, D, tTile * 16, W, D, 0, lane, c);

  const int hl = (lane >> 4) & 1;
  if (mat == 2) {           // V transposed: vt[head][hs][t]
    __bf16* out = vt + (size_t)head * HS * T;
#pragma unroll
    for (int j = 0; j < 4; ++j)
#pragma unroll
      for (int i = 0; i < 8; ++i)
        out[(size_t)(j * 16 + (lane & 15)) * T + (tTile * 16 + i + 8 * hl)] =
            f2bf(c[j][i]);
  } else {                  // Q / K row-major [head][t][hs]
    __bf16* out = (mat == 0 ? q : k) + (size_t)head * T * HS;
#pragma unroll
    for (int j = 0; j < 4; ++j)
#pragma unroll
      for (int i = 0; i < 8; ++i)
        out[(size_t)(tTile * 16 + i + 8 * hl) * HS + j * 16 + (lane & 15)] =
            f2bf(c[j][i]);
  }
}

// -------------------- Flash attention (causal, per head) -------------------
// One wave owns (head, 16-query tile); streams 32-key tiles, online softmax.
__global__ __launch_bounds__(128) void attn_kernel(const __bf16* __restrict__ Qg,
    const __bf16* __restrict__ Kg, const __bf16* __restrict__ Vt,
    __bf16* __restrict__ av) {
  const int lane  = threadIdx.x & 31;
  const int gw    = (blockIdx.x * blockDim.x + threadIdx.x) >> 5;
  const int head  = gw >> 7;          // T/16 == 128 query tiles per head
  const int qTile = gw & 127;
  const int qBase = qTile * 16;
  const int hl    = (lane >> 4) & 1;
  const __bf16* Q = Qg + (size_t)head * T * HS;
  const __bf16* K = Kg + (size_t)head * T * HS;
  const __bf16* V = Vt + (size_t)head * HS * T;   // [HS][T]

  __shared__ alignas(16) __bf16 plds[4][16 * 32];  // per-wave P staging
  __bf16* myP = plds[threadIdx.x >> 5];

  const v16bf Aq0 = load_a_bf16(Q, HS, qBase, 0,  lane);
  const v16bf Aq1 = load_a_bf16(Q, HS, qBase, 32, lane);

  float m[8], l[8];
  v8f O[4];
#pragma unroll
  for (int i = 0; i < 8; ++i) { m[i] = -__builtin_inff(); l[i] = 0.f; }
#pragma unroll
  for (int j = 0; j < 4; ++j)
#pragma unroll
    for (int i = 0; i < 8; ++i) O[j][i] = 0.f;

  const float scale = 0.125f;                      // 1/sqrt(HS)
  for (int sBase = 0; sBase < qBase + 16; sBase += 32) {
    v8f S0 = {0.f, 0.f, 0.f, 0.f, 0.f, 0.f, 0.f, 0.f};
    v8f S1 = {0.f, 0.f, 0.f, 0.f, 0.f, 0.f, 0.f, 0.f};
    S0 = wmma_bf16(Aq0, load_bT_bf16(K, HS, 0,  sBase,      lane), S0);
    S0 = wmma_bf16(Aq1, load_bT_bf16(K, HS, 32, sBase,      lane), S0);
    S1 = wmma_bf16(Aq0, load_bT_bf16(K, HS, 0,  sBase + 16, lane), S1);
    S1 = wmma_bf16(Aq1, load_bT_bf16(K, HS, 32, sBase + 16, lane), S1);
    // Hoist the V operand loads so their latency hides under the softmax VALU.
    v16bf Bv[4];
#pragma unroll
    for (int j = 0; j < 4; ++j) Bv[j] = load_bT_bf16(V, T, sBase, j * 16, lane);
    // Prefetch next key tile of K.
    __builtin_prefetch(K + (size_t)(sBase + 32 + (lane & 15)) * HS, 0, 1);

    const int col0 = sBase + (lane & 15);
#pragma unroll
    for (int i = 0; i < 8; ++i) {
      const int qrow = qBase + i + 8 * hl;
      float s0 = (col0      <= qrow) ? S0[i] * scale : -__builtin_inff();
      float s1 = (col0 + 16 <= qrow) ? S1[i] * scale : -__builtin_inff();
      float mt = fmaxf(s0, s1);
#pragma unroll
      for (int o = 1; o < 16; o <<= 1) mt = fmaxf(mt, __shfl_xor(mt, o, 16));
      const float mn = fmaxf(m[i], mt);
      const float p0 = __expf(s0 - mn);
      const float p1 = __expf(s1 - mn);
      float rs = p0 + p1;
#pragma unroll
      for (int o = 1; o < 16; o <<= 1) rs += __shfl_xor(rs, o, 16);
      const float alpha = __expf(m[i] - mn);
      l[i] = l[i] * alpha + rs;
      m[i] = mn;
#pragma unroll
      for (int j = 0; j < 4; ++j) O[j][i] *= alpha;
      // stage P into wave-private LDS (transpose into A-operand layout)
      myP[(i + 8 * hl) * 32 + (lane & 15)]      = f2bf(p0);
      myP[(i + 8 * hl) * 32 + 16 + (lane & 15)] = f2bf(p1);
    }
    // DS ops are in-order within a wave -> no barrier for wave-private LDS
    const v16bf Ap = load_a_bf16(myP, 32, 0, 0, lane);
#pragma unroll
    for (int j = 0; j < 4; ++j) O[j] = wmma_bf16(Ap, Bv[j], O[j]);
  }

#pragma unroll
  for (int i = 0; i < 8; ++i) {
    const float inv  = 1.f / l[i];
    const size_t row = (size_t)(qBase + i + 8 * hl) * (H * HS);
    const int colb   = head * HS + (lane & 15);
#pragma unroll
    for (int j = 0; j < 4; ++j)
      av[row + colb + j * 16] = f2bf(O[j][i] * inv);
  }
}

// ----------------- attention output projection + residual ------------------
// wave = tTile*(D/64) + nTile  -> 128*16 = 2048 waves
__global__ __launch_bounds__(256) void proj_kernel(const __bf16* __restrict__ av,
    const __bf16* __restrict__ WoT,   // [D(N) x D(K)] bf16
    const float* __restrict__ bo, const float* __restrict__ x,
    float* __restrict__ x1) {
  const int lane = threadIdx.x & 31;
  int gw = (blockIdx.x * blockDim.x + threadIdx.x) >> 5;
  const int nTile = gw & 15;
  const int tTile = gw >> 4;
  v8f c[4];
#pragma unroll
  for (int j = 0; j < 4; ++j)
#pragma unroll
    for (int i = 0; i < 8; ++i) c[j][i] = 0.f;

  gemm16x64<D>(av, D, tTile * 16, WoT, D, nTile * 64, lane, c);

  const int hl = (lane >> 4) & 1;
#pragma unroll
  for (int j = 0; j < 4; ++j) {
    const int col = nTile * 64 + j * 16 + (lane & 15);
#pragma unroll
    for (int i = 0; i < 8; ++i) {
      const size_t idx = (size_t)(tTile * 16 + i + 8 * hl) * D + col;
      x1[idx] = x[idx] + c[j][i] + bo[col];
    }
  }
}

// ------------------------------- FFN layer 1 -------------------------------
// wave = tTile*(DFF/64) + nTile -> 128*64 = 8192 waves
__global__ __launch_bounds__(256) void ffn1_kernel(const __bf16* __restrict__ h2,
    const __bf16* __restrict__ W1T,   // [DFF(N) x D(K)] bf16
    const float* __restrict__ b1, __bf16* __restrict__ ff1) {
  const int lane = threadIdx.x & 31;
  int gw = (blockIdx.x * blockDim.x + threadIdx.x) >> 5;
  const int nTile = gw & 63;
  const int tTile = gw >> 6;
  v8f c[4];
#pragma unroll
  for (int j = 0; j < 4; ++j)
#pragma unroll
    for (int i = 0; i < 8; ++i) c[j][i] = 0.f;

  gemm16x64<D>(h2, D, tTile * 16, W1T, D, nTile * 64, lane, c);

  const int hl = (lane >> 4) & 1;
#pragma unroll
  for (int j = 0; j < 4; ++j) {
    const int col = nTile * 64 + j * 16 + (lane & 15);
#pragma unroll
    for (int i = 0; i < 8; ++i) {
      const float t = fmaxf(c[j][i] + b1[col], 0.f);      // ReLU
      ff1[(size_t)(tTile * 16 + i + 8 * hl) * DFF + col] = f2bf(t);
    }
  }
}

// ----------------------- FFN layer 2 + final residual ----------------------
// wave = tTile*(D/64) + nTile -> 128*16 = 2048 waves, K = 4096
__global__ __launch_bounds__(256) void ffn2_kernel(const __bf16* __restrict__ ff1,
    const __bf16* __restrict__ W2T,   // [D(N) x DFF(K)] bf16
    const float* __restrict__ b2, const float* __restrict__ x1,
    float* __restrict__ out) {
  const int lane = threadIdx.x & 31;
  int gw = (blockIdx.x * blockDim.x + threadIdx.x) >> 5;
  const int nTile = gw & 15;
  const int tTile = gw >> 4;
  v8f c[4];
#pragma unroll
  for (int j = 0; j < 4; ++j)
#pragma unroll
    for (int i = 0; i < 8; ++i) c[j][i] = 0.f;

  gemm16x64<DFF>(ff1, DFF, tTile * 16, W2T, DFF, nTile * 64, lane, c);

  const int hl = (lane >> 4) & 1;
#pragma unroll
  for (int j = 0; j < 4; ++j) {
    const int col = nTile * 64 + j * 16 + (lane & 15);
#pragma unroll
    for (int i = 0; i < 8; ++i) {
      const size_t idx = (size_t)(tTile * 16 + i + 8 * hl) * D + col;
      out[idx] = x1[idx] + c[j][i] + b2[col];
    }
  }
}

// ------------------------------- launcher ----------------------------------
extern "C" void kernel_launch(void* const* d_in, const int* in_sizes, int n_in,
                              void* d_out, int out_size, void* d_ws, size_t ws_size,
                              hipStream_t stream) {
  (void)in_sizes; (void)n_in; (void)out_size; (void)ws_size;
  const float* x    = (const float*)d_in[0];
  const float* Wq   = (const float*)d_in[1];
  const float* Wk   = (const float*)d_in[2];
  const float* Wv   = (const float*)d_in[3];
  const float* Wo   = (const float*)d_in[4];
  const float* bo   = (const float*)d_in[5];
  const float* W1   = (const float*)d_in[6];
  const float* b1   = (const float*)d_in[7];
  const float* W2   = (const float*)d_in[8];
  const float* b2   = (const float*)d_in[9];
  const float* ln1s = (const float*)d_in[10];
  const float* ln1b = (const float*)d_in[11];
  const float* ln2s = (const float*)d_in[12];
  const float* ln2b = (const float*)d_in[13];

  char* ws = (char*)d_ws;
  size_t off = 0;
  auto take = [&](size_t bytes) {
    char* p = ws + off;
    off = (off + bytes + 255) & ~(size_t)255;
    return p;
  };
  __bf16* wqkvT = (__bf16*)take((size_t)3 * H * HS * D * 2);  // 6 MB
  __bf16* woT   = (__bf16*)take((size_t)D * D * 2);           // 2 MB
  __bf16* w1T   = (__bf16*)take((size_t)DFF * D * 2);         // 8 MB
  __bf16* w2T   = (__bf16*)take((size_t)D * DFF * 2);         // 8 MB
  __bf16* h1    = (__bf16*)take((size_t)T * D * 2);           // LN1 out
  __bf16* qb    = (__bf16*)take((size_t)H * T * HS * 2);      // Q
  __bf16* kb    = (__bf16*)take((size_t)H * T * HS * 2);      // K
  __bf16* vtb   = (__bf16*)take((size_t)H * HS * T * 2);      // V^T
  __bf16* avb   = (__bf16*)take((size_t)T * D * 2);           // attn out
  float*  x1    = (float*) take((size_t)T * D * 4);           // residual (f32)
  __bf16* h2    = (__bf16*)take((size_t)T * D * 2);           // LN2 out
  __bf16* ff1   = (__bf16*)take((size_t)T * DFF * 2);         // FFN hidden

  // weight prep (memory-bound, ~72 MB total traffic ≈ 3 µs at 23.3 TB/s)
  wcvt_qkv_kernel<<<12288, 256, 0, stream>>>(Wq, Wk, Wv, wqkvT);
  wcvt_t_kernel  <<<4096,  256, 0, stream>>>(Wo, woT, D,   10);
  wcvt_t_kernel  <<<16384, 256, 0, stream>>>(W1, w1T, DFF, 10);
  wcvt_t_kernel  <<<16384, 256, 0, stream>>>(W2, w2T, D,   12);

  ln_kernel  <<<T,    256, 0, stream>>>(x,  ln1s, ln1b, h1);
  qkv_kernel <<<768,  256, 0, stream>>>(h1, wqkvT, qb, kb, vtb);   // 6144 waves
  attn_kernel<<<512,  128, 0, stream>>>(qb, kb, vtb, avb);         // 2048 waves
  proj_kernel<<<256,  256, 0, stream>>>(avb, woT, bo, x, x1);      // 2048 waves
  ln_kernel  <<<T,    256, 0, stream>>>(x1, ln2s, ln2b, h2);
  ffn1_kernel<<<1024, 256, 0, stream>>>(h2, w1T, b1, ff1);         // 8192 waves
  ffn2_kernel<<<256,  256, 0, stream>>>(ff1, w2T, b2, x1, (float*)d_out);
}
